// ScatteringTransform_6262062318302
// MI455X (gfx1250) — matrix operand order repaired and checked by reading, not compiled, not run on hardware
//
#include <hip/hip_runtime.h>

typedef __attribute__((ext_vector_type(16))) __bf16 v16bf;
typedef __attribute__((ext_vector_type(8)))  __bf16 v8bf;
typedef __attribute__((ext_vector_type(8)))  float  v8f;

#define N_DIM    2048
#define PHI_COLS 21   // 1 + 4 + 16

// Convert 8 contiguous f32 -> bf16 into elements [BASE..BASE+7] of a v16bf.
template <int BASE>
__device__ inline void load8_cvt(const float* __restrict__ p, v16bf& v) {
  const float4 q0 = *reinterpret_cast<const float4*>(p);
  const float4 q1 = *reinterpret_cast<const float4*>(p + 4);
  v[BASE + 0] = (__bf16)q0.x; v[BASE + 1] = (__bf16)q0.y;
  v[BASE + 2] = (__bf16)q0.z; v[BASE + 3] = (__bf16)q0.w;
  v[BASE + 4] = (__bf16)q1.x; v[BASE + 5] = (__bf16)q1.y;
  v[BASE + 6] = (__bf16)q1.z; v[BASE + 7] = (__bf16)q1.w;
}

// Two 16B-aligned v8bf chunks -> one v16bf operand.
__device__ inline v16bf pack16bf(const __bf16* __restrict__ p0,
                                 const __bf16* __restrict__ p1) {
  const v8bf lo = *reinterpret_cast<const v8bf*>(p0);
  const v8bf hi = *reinterpret_cast<const v8bf*>(p1);
  return __builtin_shufflevector(lo, hi, 0, 1, 2, 3, 4, 5, 6, 7,
                                 8, 9, 10, 11, 12, 13, 14, 15);
}

// ---------------- bulk f32 -> bf16 conversion (8 elems/thread) ----------------
__global__ __launch_bounds__(256) void cvt_bf16_kernel(const float* __restrict__ src,
                                                       __bf16* __restrict__ dst,
                                                       int n8) {
  const int i = blockIdx.x * 256 + threadIdx.x;
  if (i < n8) {
    const float4 q0 = *reinterpret_cast<const float4*>(src + (size_t)i * 8);
    const float4 q1 = *reinterpret_cast<const float4*>(src + (size_t)i * 8 + 4);
    v8bf o;
    o[0] = (__bf16)q0.x; o[1] = (__bf16)q0.y; o[2] = (__bf16)q0.z; o[3] = (__bf16)q0.w;
    o[4] = (__bf16)q1.x; o[5] = (__bf16)q1.y; o[6] = (__bf16)q1.z; o[7] = (__bf16)q1.w;
    *reinterpret_cast<v8bf*>(dst + (size_t)i * 8) = o;
  }
}

// ---------------- phi0[b,f] = x[b,f,:] . lowpass ----------------
__global__ __launch_bounds__(256) void phi0_kernel(const float* __restrict__ x,
                                                   const float* __restrict__ lowpass,
                                                   float* __restrict__ out) {
  const int row = blockIdx.x;              // b*16+f, 128 rows
  const int tid = threadIdx.x;
  const float* xr = x + (size_t)row * N_DIM;
  float p = 0.f;
  for (int i = tid; i < N_DIM; i += 256) p += xr[i] * lowpass[i];
  __shared__ float red[256];
  red[tid] = p; __syncthreads();
  for (int s = 128; s > 0; s >>= 1) {
    if (tid < s) red[tid] += red[tid + s];
    __syncthreads();
  }
  if (tid == 0) out[(size_t)row * PHI_COLS] = red[0];
}

// ---------------- Layer 1: C = x(128x2048) @ psi_j(2048x2048) ----------------
// Wave macro-tile: RB=4 row blocks x MT=2 m-tiles (8 WMMA accumulators).
// wave-tile = (rbg*4 + j)*64 + mtp ; rbg in [0,2), 512 wave-tiles = 64 blocks.
// PRE=true: A from xb (bf16), B from psib (bf16). PRE=false: f32 + on-the-fly cvt.
template <bool PRE>
__global__ __launch_bounds__(256) void layer1_kernel(const float* __restrict__ x,
                                                     const __bf16* __restrict__ xb,
                                                     const float* __restrict__ psi,
                                                     const __bf16* __restrict__ psib,
                                                     const float* __restrict__ lowpass,
                                                     __bf16* __restrict__ s1,
                                                     float* __restrict__ part1) {
  const int wave = threadIdx.x >> 5;
  const int lane = threadIdx.x & 31;
  const int tile = blockIdx.x * 8 + wave;      // 0..511
  const int mtp  = tile & 63;                  // m-tile pair: mt = mtp*2 + mi
  const int j    = (tile >> 6) & 3;
  const int rbg  = tile >> 8;                  // 0..1 -> rows rbg*64 ..
  const int half = lane >> 4;
  const int mrow = lane & 15;

  size_t aoff[4];
#pragma unroll
  for (int rbl = 0; rbl < 4; ++rbl)
    aoff[rbl] = (size_t)(rbg * 64 + rbl * 16 + mrow) * N_DIM;
  const size_t boff = (size_t)j * N_DIM * N_DIM + (size_t)mtp * 32;

  v8f c[4][2];
#pragma unroll
  for (int rbl = 0; rbl < 4; ++rbl)
#pragma unroll
    for (int mi = 0; mi < 2; ++mi) c[rbl][mi] = (v8f)(0.f);

  for (int kb = 0; kb < N_DIM; kb += 32) {
    v16bf aV[4], bV[2];
#pragma unroll
    for (int rbl = 0; rbl < 4; ++rbl) {
      // A 16x32 bf16 layout: lane m=mrow, K chunks {0..7,16..23}/{8..15,24..31}
      if (PRE) {
        aV[rbl] = pack16bf(xb + aoff[rbl] + kb + half * 8,
                           xb + aoff[rbl] + kb + 16 + half * 8);
      } else {
        load8_cvt<0>(x + aoff[rbl] + kb + half * 8, aV[rbl]);
        load8_cvt<8>(x + aoff[rbl] + kb + 16 + half * 8, aV[rbl]);
      }
    }
#pragma unroll
    for (int mi = 0; mi < 2; ++mi) {
      // B 32x16 layout: lane n=mrow, K chunks {0..7,16..23}/{8..15,24..31}
      if (PRE) {
        const __bf16* bq = psib + boff + (size_t)(kb + half * 8) * N_DIM + mi * 16 + mrow;
        for (int s = 0; s < 16; ++s) bV[mi][s] = bq[(size_t)(s + (s >> 3) * 8) * N_DIM];
      } else {
        const float* bp = psi + boff + (size_t)(kb + half * 8) * N_DIM + mi * 16 + mrow;
        for (int s = 0; s < 16; ++s) bV[mi][s] = (__bf16)bp[(size_t)(s + (s >> 3) * 8) * N_DIM];
      }
    }
#pragma unroll
    for (int rbl = 0; rbl < 4; ++rbl)
#pragma unroll
      for (int mi = 0; mi < 2; ++mi)
        c[rbl][mi] = __builtin_amdgcn_wmma_f32_16x16x32_bf16(
            false, aV[rbl], false, bV[mi], (short)0, c[rbl][mi], false, false);
  }

  // Epilogue: abs -> S1 (bf16); lowpass-weighted partials (summed over 2 m-tiles).
  float acc[4][8];
#pragma unroll
  for (int rbl = 0; rbl < 4; ++rbl)
#pragma unroll
    for (int r = 0; r < 8; ++r) acc[rbl][r] = 0.f;

#pragma unroll
  for (int mi = 0; mi < 2; ++mi) {
    const int col = (mtp * 2 + mi) * 16 + mrow;
    const float w = lowpass[col];
#pragma unroll
    for (int rbl = 0; rbl < 4; ++rbl) {
      const int b = rbg * 4 + rbl;
#pragma unroll
      for (int r = 0; r < 8; ++r) {
        const int f = r + half * 8;              // C: M = r + half*8, N = mrow
        const float v = fabsf(c[rbl][mi][r]);
        s1[(size_t)(b * 64 + j * 16 + f) * N_DIM + col] = (__bf16)v;
        acc[rbl][r] += v * w;
      }
    }
  }
#pragma unroll
  for (int off = 1; off < 16; off <<= 1)
#pragma unroll
    for (int rbl = 0; rbl < 4; ++rbl)
#pragma unroll
      for (int r = 0; r < 8; ++r) acc[rbl][r] += __shfl_xor(acc[rbl][r], off, 32);
  if (mrow == 0) {
#pragma unroll
    for (int rbl = 0; rbl < 4; ++rbl) {
      const int b = rbg * 4 + rbl;
#pragma unroll
      for (int r = 0; r < 8; ++r)
        part1[(size_t)((b * 4 + j) * 64 + mtp) * 16 + r + half * 8] = acc[rbl][r];
    }
  }
}

// ---------------- Layer 2: C = S1(512x2048,bf16) @ psi_j; only phi partials ----
// wave-tile = (rbg*4 + j)*64 + mtp ; rbg in [0,8), 2048 wave-tiles = 256 blocks.
template <bool PRE>
__global__ __launch_bounds__(256) void layer2_kernel(const __bf16* __restrict__ s1,
                                                     const float* __restrict__ psi,
                                                     const __bf16* __restrict__ psib,
                                                     const float* __restrict__ lowpass,
                                                     float* __restrict__ part2) {
  const int wave = threadIdx.x >> 5;
  const int lane = threadIdx.x & 31;
  const int tile = blockIdx.x * 8 + wave;      // 0..2047
  const int mtp  = tile & 63;
  const int j    = (tile >> 6) & 3;
  const int rbg  = tile >> 8;                  // 0..7 -> S1 rows rbg*64 ..
  const int half = lane >> 4;
  const int mrow = lane & 15;

  size_t aoff[4];
#pragma unroll
  for (int rbl = 0; rbl < 4; ++rbl)
    aoff[rbl] = (size_t)(rbg * 64 + rbl * 16 + mrow) * N_DIM;
  const size_t boff = (size_t)j * N_DIM * N_DIM + (size_t)mtp * 32;

  v8f c[4][2];
#pragma unroll
  for (int rbl = 0; rbl < 4; ++rbl)
#pragma unroll
    for (int mi = 0; mi < 2; ++mi) c[rbl][mi] = (v8f)(0.f);

  for (int kb = 0; kb < N_DIM; kb += 32) {
    v16bf aV[4], bV[2];
#pragma unroll
    for (int rbl = 0; rbl < 4; ++rbl)
      aV[rbl] = pack16bf(s1 + aoff[rbl] + kb + half * 8,
                         s1 + aoff[rbl] + kb + 16 + half * 8);
#pragma unroll
    for (int mi = 0; mi < 2; ++mi) {
      if (PRE) {
        const __bf16* bq = psib + boff + (size_t)(kb + half * 8) * N_DIM + mi * 16 + mrow;
        for (int s = 0; s < 16; ++s) bV[mi][s] = bq[(size_t)(s + (s >> 3) * 8) * N_DIM];
      } else {
        const float* bp = psi + boff + (size_t)(kb + half * 8) * N_DIM + mi * 16 + mrow;
        for (int s = 0; s < 16; ++s) bV[mi][s] = (__bf16)bp[(size_t)(s + (s >> 3) * 8) * N_DIM];
      }
    }
#pragma unroll
    for (int rbl = 0; rbl < 4; ++rbl)
#pragma unroll
      for (int mi = 0; mi < 2; ++mi)
        c[rbl][mi] = __builtin_amdgcn_wmma_f32_16x16x32_bf16(
            false, aV[rbl], false, bV[mi], (short)0, c[rbl][mi], false, false);
  }

  float acc[4][8];
#pragma unroll
  for (int rbl = 0; rbl < 4; ++rbl)
#pragma unroll
    for (int r = 0; r < 8; ++r) acc[rbl][r] = 0.f;

#pragma unroll
  for (int mi = 0; mi < 2; ++mi) {
    const float w = lowpass[(mtp * 2 + mi) * 16 + mrow];
#pragma unroll
    for (int rbl = 0; rbl < 4; ++rbl)
#pragma unroll
      for (int r = 0; r < 8; ++r) acc[rbl][r] += fabsf(c[rbl][mi][r]) * w;
  }
#pragma unroll
  for (int off = 1; off < 16; off <<= 1)
#pragma unroll
    for (int rbl = 0; rbl < 4; ++rbl)
#pragma unroll
      for (int r = 0; r < 8; ++r) acc[rbl][r] += __shfl_xor(acc[rbl][r], off, 32);
  if (mrow == 0) {
#pragma unroll
    for (int rbl = 0; rbl < 4; ++rbl) {
      const int rb2 = rbg * 4 + rbl;           // b*4 + k1
#pragma unroll
      for (int r = 0; r < 8; ++r)
        part2[(size_t)((rb2 * 4 + j) * 64 + mtp) * 16 + r + half * 8] = acc[rbl][r];
    }
  }
}

// ---------------- Deterministic reductions over the 64 m-tile-pairs ----------------
__global__ __launch_bounds__(64) void reduce1_kernel(const float* __restrict__ part1,
                                                     float* __restrict__ out) {
  const int o = blockIdx.x;                 // (b*4+j)*16 + f, 512 outputs
  const int f = o & 15;
  const int bj = o >> 4;                    // b*4 + j
  const int b = bj >> 2, j = bj & 3;
  const int t = threadIdx.x;                // mtp
  float v = part1[(size_t)(bj * 64 + t) * 16 + f];
  __shared__ float red[64];
  red[t] = v; __syncthreads();
  for (int s = 32; s > 0; s >>= 1) {
    if (t < s) red[t] += red[t + s];
    __syncthreads();
  }
  if (t == 0) out[(size_t)(b * 16 + f) * PHI_COLS + 1 + j] = red[0];
}

__global__ __launch_bounds__(64) void reduce2_kernel(const float* __restrict__ part2,
                                                     float* __restrict__ out) {
  const int o = blockIdx.x;                 // idx*16 + f, idx = b*16+k1*4+j; 2048 outputs
  const int f = o & 15;
  const int idx = o >> 4;                   // 0..127  (== rb2*4 + j)
  const int b = idx >> 4, k1 = (idx >> 2) & 3, j = idx & 3;
  const int t = threadIdx.x;
  float v = part2[(size_t)(idx * 64 + t) * 16 + f];
  __shared__ float red[64];
  red[t] = v; __syncthreads();
  for (int s = 32; s > 0; s >>= 1) {
    if (t < s) red[t] += red[t + s];
    __syncthreads();
  }
  if (t == 0) out[(size_t)(b * 16 + f) * PHI_COLS + 5 + k1 * 4 + j] = red[0];
}

extern "C" void kernel_launch(void* const* d_in, const int* in_sizes, int n_in,
                              void* d_out, int out_size, void* d_ws, size_t ws_size,
                              hipStream_t stream) {
  (void)in_sizes; (void)n_in; (void)out_size;
  const float* x       = (const float*)d_in[0];
  const float* psi     = (const float*)d_in[1];
  const float* lowpass = (const float*)d_in[2];
  float* out = (float*)d_out;

  // Common workspace (always needed):
  //   s1    : 512 x 2048 bf16                     = 2 MB
  //   part1 : 32*64*16 f32                        = 128 KB
  //   part2 : 128*64*16 f32                       = 512 KB
  // Optional pre-converted operands (bf16):
  //   psib  : 4*2048*2048 bf16                    = 32 MB
  //   xb    : 128*2048 bf16                       = 512 KB
  const size_t s1_bytes    = (size_t)512 * N_DIM * 2;
  const size_t part1_elems = (size_t)32 * 64 * 16;
  const size_t part2_elems = (size_t)128 * 64 * 16;
  const size_t base_bytes  = s1_bytes + (part1_elems + part2_elems) * 4;
  const size_t psib_bytes  = (size_t)4 * N_DIM * N_DIM * 2;
  const size_t xb_bytes    = (size_t)128 * N_DIM * 2;

  char* ws = (char*)d_ws;
  __bf16* s1   = (__bf16*)ws;
  float* part1 = (float*)(ws + s1_bytes);
  float* part2 = part1 + part1_elems;
  __bf16* psib = (__bf16*)(ws + base_bytes);
  __bf16* xb   = (__bf16*)(ws + base_bytes + psib_bytes);

  // ws_size is fixed for the lifetime of the harness -> branch is deterministic.
  const bool pre = ws_size >= base_bytes + psib_bytes + xb_bytes;

  phi0_kernel<<<128, 256, 0, stream>>>(x, lowpass, out);
  if (pre) {
    const int psi_n8 = 4 * N_DIM * N_DIM / 8;   // 2,097,152
    const int x_n8   = 128 * N_DIM / 8;         // 32,768
    cvt_bf16_kernel<<<(psi_n8 + 255) / 256, 256, 0, stream>>>(psi, psib, psi_n8);
    cvt_bf16_kernel<<<(x_n8 + 255) / 256, 256, 0, stream>>>(x, xb, x_n8);
    layer1_kernel<true><<<64, 256, 0, stream>>>(x, xb, psi, psib, lowpass, s1, part1);
    reduce1_kernel<<<512, 64, 0, stream>>>(part1, out);
    layer2_kernel<true><<<256, 256, 0, stream>>>(s1, psi, psib, lowpass, part2);
    reduce2_kernel<<<2048, 64, 0, stream>>>(part2, out);
  } else {
    layer1_kernel<false><<<64, 256, 0, stream>>>(x, nullptr, psi, nullptr, lowpass, s1, part1);
    reduce1_kernel<<<512, 64, 0, stream>>>(part1, out);
    layer2_kernel<false><<<256, 256, 0, stream>>>(s1, psi, nullptr, lowpass, part2);
    reduce2_kernel<<<2048, 64, 0, stream>>>(part2, out);
  }
}
